// GAT_60146722013607
// MI455X (gfx1250) — matrix operand order repaired and checked by reading, not compiled, not run on hardware
//
#include <hip/hip_runtime.h>
#include <hip/hip_bf16.h>

// ---------- CDNA5 WMMA vector types ----------
typedef __attribute__((ext_vector_type(16))) __bf16 v16bf;
typedef __attribute__((ext_vector_type(8)))  __bf16 v8bf;
typedef __attribute__((ext_vector_type(8)))  float  v8f;
typedef __attribute__((ext_vector_type(4)))  float  f32x4;

#define F_IN 512
#define H1 8
#define C1 8
#define HC1 64            // H1*C1
#define NCLS 40
#define NPAD 48           // NCLS padded to 3 WMMA n-tiles
#define NEG_SLOPE 0.2f

// ---------------- utility kernels ----------------
__global__ void fill_u32_kernel(unsigned int* __restrict__ p, long long n, unsigned int v) {
    long long i = (long long)blockIdx.x * blockDim.x + threadIdx.x;
    if (i < n) p[i] = v;
}

// W1 [512,64] f32 row-major -> W1bfT [64][512] bf16 (transposed, contiguous in K)
__global__ void conv_w1_kernel(const float* __restrict__ W1, __bf16* __restrict__ Wt) {
    int t = blockIdx.x * blockDim.x + threadIdx.x;     // t = k*64 + n
    if (t >= F_IN * HC1) return;
    int k = t >> 6, n = t & 63;
    Wt[(size_t)n * F_IN + k] = (__bf16)W1[(size_t)k * HC1 + n];
}

// W2 [64,40] f32 -> W2bfT [48][64] bf16 (transposed, zero-padded cols 40..47)
__global__ void conv_w2_kernel(const float* __restrict__ W2, __bf16* __restrict__ Wt) {
    int t = blockIdx.x * blockDim.x + threadIdx.x;     // t = k*48 + n
    if (t >= HC1 * NPAD) return;
    int k = t / NPAD, n = t % NPAD;
    float v = (n < NCLS) ? W2[(size_t)k * NCLS + n] : 0.f;
    Wt[(size_t)n * HC1 + k] = (__bf16)v;
}

// ---------------- GEMM1: h1 = x @ W1  (M=N_nodes, K=512, N=64), bf16 WMMA ----------------
// A-fragment (16x32 bf16, wave32): lane m = lane&15; per lane the 16 elements are
// contiguous: k in [k0+16*half, +16).
// B-fragment (32x16): lane n = lane&15; 16 contiguous elements at k0+16*half of Wt[n][:].
__global__ __launch_bounds__(256) void gemm1_kernel(const float* __restrict__ x,
                                                    const __bf16* __restrict__ Wt,
                                                    float* __restrict__ h1, int Nn) {
    const int wave = threadIdx.x >> 5;
    const int lane = threadIdx.x & 31;
    const int half = lane >> 4;
    const int l16  = lane & 15;
    const int rowBase = blockIdx.x * 128 + wave * 16;
    const int arow = rowBase + l16;
    // clamped pointer: loads always safe; invalid rows only corrupt their own D rows,
    // which the store guard drops.
    const float* xa = x + (size_t)(arow < Nn ? arow : 0) * F_IN + 16 * half;
    const __bf16* w0 = Wt + (size_t)(l16 +  0) * F_IN + 16 * half;
    const __bf16* w1 = Wt + (size_t)(l16 + 16) * F_IN + 16 * half;
    const __bf16* w2 = Wt + (size_t)(l16 + 32) * F_IN + 16 * half;
    const __bf16* w3 = Wt + (size_t)(l16 + 48) * F_IN + 16 * half;

    v8f c0 = {}, c1 = {}, c2 = {}, c3 = {};

    for (int k0 = 0; k0 < F_IN; k0 += 32) {
        __builtin_prefetch(xa + k0 + 128, 0, 3);      // global_prefetch_b8
        f32x4 f0 = *(const f32x4*)(xa + k0);
        f32x4 f1 = *(const f32x4*)(xa + k0 + 4);
        f32x4 f2 = *(const f32x4*)(xa + k0 + 8);
        f32x4 f3 = *(const f32x4*)(xa + k0 + 12);
        v16bf a;
        #pragma unroll
        for (int i = 0; i < 4; ++i) {
            a[i]      = (__bf16)f0[i];
            a[i + 4]  = (__bf16)f1[i];
            a[i + 8]  = (__bf16)f2[i];
            a[i + 12] = (__bf16)f3[i];
        }
        v16bf b0 = *(const v16bf*)(w0 + k0);
        v16bf b1 = *(const v16bf*)(w1 + k0);
        v16bf b2 = *(const v16bf*)(w2 + k0);
        v16bf b3 = *(const v16bf*)(w3 + k0);
        c0 = __builtin_amdgcn_wmma_f32_16x16x32_bf16(false, a, false, b0, (short)0, c0, false, false);
        c1 = __builtin_amdgcn_wmma_f32_16x16x32_bf16(false, a, false, b1, (short)0, c1, false, false);
        c2 = __builtin_amdgcn_wmma_f32_16x16x32_bf16(false, a, false, b2, (short)0, c2, false, false);
        c3 = __builtin_amdgcn_wmma_f32_16x16x32_bf16(false, a, false, b3, (short)0, c3, false, false);
    }

    #pragma unroll
    for (int r = 0; r < 8; ++r) {
        int row = rowBase + half * 8 + r;
        if (row < Nn) {
            float* hp = h1 + (size_t)row * HC1;
            hp[ 0 + l16] = c0[r];
            hp[16 + l16] = c1[r];
            hp[32 + l16] = c2[r];
            hp[48 + l16] = c3[r];
        }
    }
}

// ---------------- per-node attention projections, layer 1 ----------------
__global__ void alpha1_kernel(const float* __restrict__ h1, const float* __restrict__ a_s,
                              const float* __restrict__ a_d, float* __restrict__ as1,
                              float* __restrict__ ad1, int Nn) {
    int t = blockIdx.x * blockDim.x + threadIdx.x;     // t = n*8 + head
    if (t >= Nn * H1) return;
    int hd = t & 7, n = t >> 3;
    const float* hp = h1 + (size_t)n * HC1 + hd * C1;
    float s = 0.f, d = 0.f;
    #pragma unroll
    for (int c = 0; c < C1; ++c) {
        float v = hp[c];
        s += v * a_s[hd * C1 + c];
        d += v * a_d[hd * C1 + c];
    }
    as1[t] = s;
    ad1[t] = d;
}

// ---------------- float atomic max via int/uint atomics ----------------
__device__ __forceinline__ void atomicMaxF(float* addr, float v) {
    if (v >= 0.f) atomicMax((int*)addr, __float_as_int(v));
    else          atomicMin((unsigned int*)addr, __float_as_uint(v));
}

__device__ __forceinline__ void edge_pair(const long long* ei, long long eid, long long E,
                                          int& src, int& dst) {
    if (eid < E) { src = (int)ei[eid]; dst = (int)ei[E + eid]; }
    else         { src = dst = (int)(eid - E); }         // self loops appended
}

// ---------------- layer-1 edge passes ----------------
__global__ void edge_max1_kernel(const long long* __restrict__ ei, const float* __restrict__ as1,
                                 const float* __restrict__ ad1, float* __restrict__ m1,
                                 long long E, int Nn) {
    long long t = (long long)blockIdx.x * blockDim.x + threadIdx.x;
    long long total = (E + Nn) * (long long)H1;
    if (t >= total) return;
    int hd = (int)(t & 7);
    long long eid = t >> 3;
    int src, dst; edge_pair(ei, eid, E, src, dst);
    float e = as1[(size_t)src * H1 + hd] + ad1[(size_t)dst * H1 + hd];
    e = (e > 0.f) ? e : NEG_SLOPE * e;
    atomicMaxF(&m1[(size_t)dst * H1 + hd], e);
}

__global__ void edge_sum1_kernel(const long long* __restrict__ ei, const float* __restrict__ as1,
                                 const float* __restrict__ ad1, const float* __restrict__ m1,
                                 float* __restrict__ s1, long long E, int Nn) {
    long long t = (long long)blockIdx.x * blockDim.x + threadIdx.x;
    long long total = (E + Nn) * (long long)H1;
    if (t >= total) return;
    int hd = (int)(t & 7);
    long long eid = t >> 3;
    int src, dst; edge_pair(ei, eid, E, src, dst);
    float e = as1[(size_t)src * H1 + hd] + ad1[(size_t)dst * H1 + hd];
    e = (e > 0.f) ? e : NEG_SLOPE * e;
    atomicAdd(&s1[(size_t)dst * H1 + hd], __expf(e - m1[(size_t)dst * H1 + hd]));
}

__global__ void edge_agg1_kernel(const long long* __restrict__ ei, const float* __restrict__ as1,
                                 const float* __restrict__ ad1, const float* __restrict__ m1,
                                 const float* __restrict__ s1, const float* __restrict__ h1,
                                 float* __restrict__ out1, long long E, int Nn) {
    long long t = (long long)blockIdx.x * blockDim.x + threadIdx.x;
    long long total = (E + Nn) * (long long)HC1;
    if (t >= total) return;
    int j = (int)(t & 63);
    int hd = j >> 3;
    long long eid = t >> 6;
    int src, dst; edge_pair(ei, eid, E, src, dst);
    float e = as1[(size_t)src * H1 + hd] + ad1[(size_t)dst * H1 + hd];
    e = (e > 0.f) ? e : NEG_SLOPE * e;
    float alpha = __expf(e - m1[(size_t)dst * H1 + hd]) / (s1[(size_t)dst * H1 + hd] + 1e-16f);
    atomicAdd(&out1[(size_t)dst * HC1 + j], h1[(size_t)src * HC1 + j] * alpha);
}

// ---------------- bias + ELU -> bf16 activations ----------------
__global__ void elu_kernel(const float* __restrict__ out1, const float* __restrict__ b1,
                           __bf16* __restrict__ h2bf, int Nn) {
    long long t = (long long)blockIdx.x * blockDim.x + threadIdx.x;
    long long total = (long long)Nn * HC1;
    if (t >= total) return;
    int j = (int)(t & 63);
    float v = out1[t] + b1[j];
    v = (v > 0.f) ? v : (__expf(v) - 1.f);
    h2bf[t] = (__bf16)v;
}

// ---------------- GEMM2: hl2 = h2 @ W2 (M=N_nodes, K=64, N=48 padded), bf16 WMMA ----------------
__global__ __launch_bounds__(256) void gemm2_kernel(const __bf16* __restrict__ h2,
                                                    const __bf16* __restrict__ Wt,
                                                    float* __restrict__ hl2, int Nn) {
    const int wave = threadIdx.x >> 5;
    const int lane = threadIdx.x & 31;
    const int half = lane >> 4;
    const int l16  = lane & 15;
    const int rowBase = blockIdx.x * 128 + wave * 16;
    const int arow = rowBase + l16;
    const __bf16* ha = h2 + (size_t)(arow < Nn ? arow : 0) * HC1 + 16 * half;
    const __bf16* w0 = Wt + (size_t)(l16 +  0) * HC1 + 16 * half;
    const __bf16* w1 = Wt + (size_t)(l16 + 16) * HC1 + 16 * half;
    const __bf16* w2 = Wt + (size_t)(l16 + 32) * HC1 + 16 * half;

    v8f c0 = {}, c1 = {}, c2 = {};

    #pragma unroll
    for (int k0 = 0; k0 < HC1; k0 += 32) {
        v8bf a_lo = *(const v8bf*)(ha + k0);
        v8bf a_hi = *(const v8bf*)(ha + k0 + 8);
        v16bf a;
        #pragma unroll
        for (int i = 0; i < 8; ++i) {
            a[i]     = a_lo[i];
            a[i + 8] = a_hi[i];
        }
        v16bf b0 = *(const v16bf*)(w0 + k0);
        v16bf b1 = *(const v16bf*)(w1 + k0);
        v16bf b2 = *(const v16bf*)(w2 + k0);
        c0 = __builtin_amdgcn_wmma_f32_16x16x32_bf16(false, a, false, b0, (short)0, c0, false, false);
        c1 = __builtin_amdgcn_wmma_f32_16x16x32_bf16(false, a, false, b1, (short)0, c1, false, false);
        c2 = __builtin_amdgcn_wmma_f32_16x16x32_bf16(false, a, false, b2, (short)0, c2, false, false);
    }

    #pragma unroll
    for (int r = 0; r < 8; ++r) {
        int row = rowBase + half * 8 + r;
        if (row < Nn) {
            float* hp = hl2 + (size_t)row * NPAD;
            hp[ 0 + l16] = c0[r];
            hp[16 + l16] = c1[r];
            hp[32 + l16] = c2[r];
        }
    }
}

// ---------------- per-node projections, layer 2 (heads=1) ----------------
__global__ void alpha2_kernel(const float* __restrict__ hl2, const float* __restrict__ a_s,
                              const float* __restrict__ a_d, float* __restrict__ as2,
                              float* __restrict__ ad2, int Nn) {
    int n = blockIdx.x * blockDim.x + threadIdx.x;
    if (n >= Nn) return;
    const float* hp = hl2 + (size_t)n * NPAD;
    float s = 0.f, d = 0.f;
    #pragma unroll
    for (int c = 0; c < NCLS; ++c) {
        float v = hp[c];
        s += v * a_s[c];
        d += v * a_d[c];
    }
    as2[n] = s;
    ad2[n] = d;
}

// ---------------- layer-2 edge passes (heads=1) ----------------
__global__ void edge_max2_kernel(const long long* __restrict__ ei, const float* __restrict__ as2,
                                 const float* __restrict__ ad2, float* __restrict__ m2,
                                 long long E, int Nn) {
    long long eid = (long long)blockIdx.x * blockDim.x + threadIdx.x;
    if (eid >= E + Nn) return;
    int src, dst; edge_pair(ei, eid, E, src, dst);
    float e = as2[src] + ad2[dst];
    e = (e > 0.f) ? e : NEG_SLOPE * e;
    atomicMaxF(&m2[dst], e);
}

__global__ void edge_sum2_kernel(const long long* __restrict__ ei, const float* __restrict__ as2,
                                 const float* __restrict__ ad2, const float* __restrict__ m2,
                                 float* __restrict__ s2, long long E, int Nn) {
    long long eid = (long long)blockIdx.x * blockDim.x + threadIdx.x;
    if (eid >= E + Nn) return;
    int src, dst; edge_pair(ei, eid, E, src, dst);
    float e = as2[src] + ad2[dst];
    e = (e > 0.f) ? e : NEG_SLOPE * e;
    atomicAdd(&s2[dst], __expf(e - m2[dst]));
}

__global__ void edge_agg2_kernel(const long long* __restrict__ ei, const float* __restrict__ as2,
                                 const float* __restrict__ ad2, const float* __restrict__ m2,
                                 const float* __restrict__ s2, const float* __restrict__ hl2,
                                 float* __restrict__ out2, long long E, int Nn) {
    long long t = (long long)blockIdx.x * blockDim.x + threadIdx.x;
    long long total = (E + Nn) * (long long)NCLS;
    if (t >= total) return;
    int c = (int)(t % NCLS);
    long long eid = t / NCLS;
    int src, dst; edge_pair(ei, eid, E, src, dst);
    float e = as2[src] + ad2[dst];
    e = (e > 0.f) ? e : NEG_SLOPE * e;
    float alpha = __expf(e - m2[dst]) / (s2[dst] + 1e-16f);
    atomicAdd(&out2[(size_t)dst * NCLS + c], hl2[(size_t)src * NPAD + c] * alpha);
}

// ---------------- bias + log_softmax ----------------
__global__ void logsoftmax_kernel(const float* __restrict__ out2, const float* __restrict__ b2,
                                  float* __restrict__ out, int Nn) {
    int n = blockIdx.x * blockDim.x + threadIdx.x;
    if (n >= Nn) return;
    const float* p = out2 + (size_t)n * NCLS;
    float v[NCLS];
    float mx = -3.402823466e+38f;
    #pragma unroll
    for (int c = 0; c < NCLS; ++c) {
        v[c] = p[c] + b2[c];
        mx = fmaxf(mx, v[c]);
    }
    float sum = 0.f;
    #pragma unroll
    for (int c = 0; c < NCLS; ++c) sum += __expf(v[c] - mx);
    float lse = mx + __logf(sum);
    float* o = out + (size_t)n * NCLS;
    #pragma unroll
    for (int c = 0; c < NCLS; ++c) o[c] = v[c] - lse;
}

// ---------------- host launcher ----------------
static inline unsigned int cdiv_ll(long long a, int b) { return (unsigned int)((a + b - 1) / b); }

extern "C" void kernel_launch(void* const* d_in, const int* in_sizes, int n_in,
                              void* d_out, int out_size, void* d_ws, size_t ws_size,
                              hipStream_t stream) {
    const float*     x   = (const float*)d_in[0];
    const long long* ei  = (const long long*)d_in[1];
    const float*     W1  = (const float*)d_in[2];
    const float*     aS1 = (const float*)d_in[3];
    const float*     aD1 = (const float*)d_in[4];
    const float*     b1  = (const float*)d_in[5];
    const float*     W2  = (const float*)d_in[6];
    const float*     aS2 = (const float*)d_in[7];
    const float*     aD2 = (const float*)d_in[8];
    const float*     b2  = (const float*)d_in[9];
    float*           out = (float*)d_out;

    const int Nn = in_sizes[0] / F_IN;
    const long long E = in_sizes[1] / 2;

    // workspace carve-out (256-byte aligned slices)
    char* ws = (char*)d_ws;
    size_t off = 0;
    auto take = [&](size_t bytes) -> char* {
        char* p = ws + off;
        off += (bytes + 255) & ~(size_t)255;
        return p;
    };
    __bf16* W1bfT = (__bf16*)take((size_t)HC1 * F_IN * 2);
    __bf16* W2bfT = (__bf16*)take((size_t)NPAD * HC1 * 2);
    float*  h1    = (float*) take((size_t)Nn * HC1 * 4);
    float*  as1   = (float*) take((size_t)Nn * H1 * 4);
    float*  ad1   = (float*) take((size_t)Nn * H1 * 4);
    float*  m1    = (float*) take((size_t)Nn * H1 * 4);
    float*  s1    = (float*) take((size_t)Nn * H1 * 4);
    float*  out1  = (float*) take((size_t)Nn * HC1 * 4);
    __bf16* h2bf  = (__bf16*)take((size_t)Nn * HC1 * 2);
    float*  hl2   = (float*) take((size_t)Nn * NPAD * 4);
    float*  as2   = (float*) take((size_t)Nn * 4);
    float*  ad2   = (float*) take((size_t)Nn * 4);
    float*  m2    = (float*) take((size_t)Nn * 4);
    float*  s2    = (float*) take((size_t)Nn * 4);
    float*  out2  = (float*) take((size_t)Nn * NCLS * 4);

    const int B = 256;
    const unsigned int NEG_INF = 0xFF800000u;

    // init accumulators
    fill_u32_kernel<<<cdiv_ll((long long)Nn * H1, B), B, 0, stream>>>((unsigned int*)m1, (long long)Nn * H1, NEG_INF);
    fill_u32_kernel<<<cdiv_ll((long long)Nn * H1, B), B, 0, stream>>>((unsigned int*)s1, (long long)Nn * H1, 0u);
    fill_u32_kernel<<<cdiv_ll((long long)Nn * HC1, B), B, 0, stream>>>((unsigned int*)out1, (long long)Nn * HC1, 0u);
    fill_u32_kernel<<<cdiv_ll((long long)Nn, B), B, 0, stream>>>((unsigned int*)m2, (long long)Nn, NEG_INF);
    fill_u32_kernel<<<cdiv_ll((long long)Nn, B), B, 0, stream>>>((unsigned int*)s2, (long long)Nn, 0u);
    fill_u32_kernel<<<cdiv_ll((long long)Nn * NCLS, B), B, 0, stream>>>((unsigned int*)out2, (long long)Nn * NCLS, 0u);

    // weight conversion (fp32 -> bf16, transposed)
    conv_w1_kernel<<<cdiv_ll((long long)F_IN * HC1, B), B, 0, stream>>>(W1, W1bfT);
    conv_w2_kernel<<<cdiv_ll((long long)HC1 * NPAD, B), B, 0, stream>>>(W2, W2bfT);

    // layer 1
    gemm1_kernel<<<(Nn + 127) / 128, B, 0, stream>>>(x, W1bfT, h1, Nn);
    alpha1_kernel<<<cdiv_ll((long long)Nn * H1, B), B, 0, stream>>>(h1, aS1, aD1, as1, ad1, Nn);
    edge_max1_kernel<<<cdiv_ll((E + Nn) * H1, B), B, 0, stream>>>(ei, as1, ad1, m1, E, Nn);
    edge_sum1_kernel<<<cdiv_ll((E + Nn) * H1, B), B, 0, stream>>>(ei, as1, ad1, m1, s1, E, Nn);
    edge_agg1_kernel<<<cdiv_ll((E + Nn) * HC1, B), B, 0, stream>>>(ei, as1, ad1, m1, s1, h1, out1, E, Nn);
    elu_kernel<<<cdiv_ll((long long)Nn * HC1, B), B, 0, stream>>>(out1, b1, h2bf, Nn);

    // layer 2
    gemm2_kernel<<<(Nn + 127) / 128, B, 0, stream>>>(h2bf, W2bfT, hl2, Nn);
    alpha2_kernel<<<cdiv_ll((long long)Nn, B), B, 0, stream>>>(hl2, aS2, aD2, as2, ad2, Nn);
    edge_max2_kernel<<<cdiv_ll(E + Nn, B), B, 0, stream>>>(ei, as2, ad2, m2, E, Nn);
    edge_sum2_kernel<<<cdiv_ll(E + Nn, B), B, 0, stream>>>(ei, as2, ad2, m2, s2, E, Nn);
    edge_agg2_kernel<<<cdiv_ll((E + Nn) * NCLS, B), B, 0, stream>>>(ei, as2, ad2, m2, s2, hl2, out2, E, Nn);

    // output
    logsoftmax_kernel<<<cdiv_ll((long long)Nn, B), B, 0, stream>>>(out2, b2, out, Nn);
}
